// QuantumAttention_41102837023077
// MI455X (gfx1250) — matrix operand order repaired and checked by reading, not compiled, not run on hardware
//
#include <hip/hip_runtime.h>
#include <hip/hip_bf16.h>
#include <math.h>

// ---------------- types / constants ----------------
typedef __attribute__((ext_vector_type(16))) __bf16 v16bf;
typedef __attribute__((ext_vector_type(8)))  float  v8f;
typedef __attribute__((ext_vector_type(4)))  unsigned int u32x4;

static constexpr int BB = 4, SS = 1024, EE = 1024, HH = 32, DD = 32;
static constexpr int TOK = BB * SS;                       // 4096 token rows
static constexpr float SCALE = 0.17677669529663689f;      // 1/sqrt(D)

struct Frag2 { u32x4 lo, hi; };

static __device__ __forceinline__ v16bf make_frag(u32x4 lo, u32x4 hi) {
  Frag2 f{lo, hi};
  return __builtin_bit_cast(v16bf, f);
}

static __device__ __forceinline__ unsigned short f2bf(float x) {
  unsigned int u = __builtin_bit_cast(unsigned int, x);
  u += 0x7fffu + ((u >> 16) & 1u);      // round-to-nearest-even
  return (unsigned short)(u >> 16);
}

static __device__ __forceinline__ v8f wmma_bf16(v16bf a, v16bf b, v8f c) {
  // D = A(16x32 bf16) x B(32x16 bf16) + C(16x16 f32)
  return __builtin_amdgcn_wmma_f32_16x16x32_bf16(false, a, false, b, (short)0, c,
                                                 false, false);
}

// CDNA5 async global->LDS copy (ASYNCcnt-tracked), GV addressing mode.
// lds_off: wave-relative LDS byte offset (low 32 bits of flat shared address,
// per ISA aperture rule).
static __device__ __forceinline__ void async_load_b128(unsigned lds_off,
                                                       const void* gaddr) {
  unsigned long long a = (unsigned long long)gaddr;
  asm volatile("global_load_async_to_lds_b128 %0, %1, off"
               :: "v"(lds_off), "v"(a) : "memory");
}
static __device__ __forceinline__ void wait_async0() {
  asm volatile("s_wait_asynccnt 0x0" ::: "memory");
}

// ---------------- elementwise / small kernels ----------------
__global__ void k_cast_bf16(const float* __restrict__ src,
                            unsigned short* __restrict__ dst, int n) {
  int i = (blockIdx.x * blockDim.x + threadIdx.x) * 4;
  if (i + 3 < n) {
    float4 v = *(const float4*)(src + i);
    unsigned long long p = (unsigned long long)f2bf(v.x)
                         | ((unsigned long long)f2bf(v.y) << 16)
                         | ((unsigned long long)f2bf(v.z) << 32)
                         | ((unsigned long long)f2bf(v.w) << 48);
    *(unsigned long long*)(dst + i) = p;
  }
}

__global__ void k_colmean(const float* __restrict__ x, float* __restrict__ xm) {
  int t = blockIdx.x * blockDim.x + threadIdx.x;          // [0, BB*EE)
  if (t >= BB * EE) return;
  int b = t / EE, e = t % EE;
  const float* p = x + (size_t)b * SS * EE + e;
  float s = 0.f;
  for (int i = 0; i < SS; ++i) s += p[(size_t)i * EE];
  xm[t] = s * (1.0f / SS);
}

// mode 0: relu, mode 1: cos
__global__ void k_dense_act(const float* __restrict__ in, const float* __restrict__ w,
                            const float* __restrict__ bias, float* __restrict__ out,
                            int M, int N, int K, int mode) {
  int t = blockIdx.x * blockDim.x + threadIdx.x;
  if (t >= M * N) return;
  int m = t / N, n = t % N;
  const float* a = in + (size_t)m * K;
  const float* b = w + (size_t)n * K;
  float s = bias ? bias[n] : 0.f;
  for (int k = 0; k < K; ++k) s += a[k] * b[k];
  out[t] = (mode == 0) ? fmaxf(s, 0.f) : cosf(s);
}

__global__ void k_add_cast(const float* __restrict__ a, const float* __restrict__ b,
                           unsigned short* __restrict__ dst, int n) {
  int i = blockIdx.x * blockDim.x + threadIdx.x;
  if (i < n) dst[i] = f2bf(a[i] + b[i]);
}

// ---------------- WMMA GEMM: C(M,N) = A(M,K)bf16 @ W(N,K)^T bf16 + bias ----------------
// Weight tile (64 rows x 32 k = 4KB) is staged into LDS once per block per k-step
// via global_load_async_to_lds_b128 (shared by all 4 waves), A-fragments are loaded
// directly from global (unique per wave).
// EPI 0: f32 row-major out
// EPI 1: bf16 heads layout (B,H,S,D), optional phase[b*DD+d] multiply
// EPI 2: bf16 transposed-V layout (B,H,D,S)
template<int EPI>
__global__ __launch_bounds__(128)
void k_gemm(const unsigned short* __restrict__ A, const unsigned short* __restrict__ W,
            const float* __restrict__ bias, float* __restrict__ outF,
            unsigned short* __restrict__ outB, const float* __restrict__ phase) {
  constexpr int K = EE;
  __shared__ unsigned short wtile[64 * 32];               // 4KB weight stage
  const unsigned wbase = (unsigned)(unsigned long long)(const void*)wtile;

  const int lane = threadIdx.x & 31;
  const int wave = threadIdx.x >> 5;
  const int m0 = blockIdx.x * 64 + wave * 16;             // 4 waves x 16 rows
  const int n0 = blockIdx.y * 64;                         // 4 n-tiles of 16

  // A fragment addressing (16-bit A 16x32): lane<16 -> K {c..c+7, c+16..c+23}, c=0; lane>=16 -> c=8
  const int arow  = m0 + (lane & 15);
  const int akoff = (lane >> 4) * 8;
  // B fragment addressing: lane holds column (lane&15) = weight row, K {c..c+7, c+16..c+23}, c=(lane>>4)*8
  const int bcol  = lane & 15;
  const int bkoff = (lane >> 4) * 8;

  // per-thread async chunk ids: 256 x 16B chunks cover the 64x32 tile
  const int id0 = threadIdx.x, id1 = threadIdx.x + 128;
  const int r0 = id0 >> 2, c0 = id0 & 3;
  const int r1 = id1 >> 2, c1 = id1 & 3;

  v8f acc[4] = {};

  const unsigned short* aptr = A + (size_t)arow * K + akoff;
  for (int k0 = 0; k0 < K; k0 += 32) {
    __syncthreads();                                      // prior reads of wtile done
    async_load_b128(wbase + (unsigned)(r0 * 64 + c0 * 16),
                    W + (size_t)(n0 + r0) * K + k0 + c0 * 8);
    async_load_b128(wbase + (unsigned)(r1 * 64 + c1 * 16),
                    W + (size_t)(n0 + r1) * K + k0 + c1 * 8);
    __builtin_prefetch(aptr + k0 + 64, 0, 0);             // global_prefetch_b8 (next A tile)
    v16bf af = make_frag(*(const u32x4*)(aptr + k0), *(const u32x4*)(aptr + k0 + 16));
    wait_async0();                                        // s_wait_asynccnt 0
    __syncthreads();                                      // all lanes' chunks visible
#pragma unroll
    for (int nt = 0; nt < 4; ++nt) {
      const unsigned short* wl = wtile + (nt * 16 + bcol) * 32 + bkoff;
      v16bf bf = make_frag(*(const u32x4*)wl, *(const u32x4*)(wl + 16));
      acc[nt] = wmma_bf16(af, bf, acc[nt]);
    }
  }

  const int mrow = (lane >> 4) * 8;
#pragma unroll
  for (int nt = 0; nt < 4; ++nt) {
    const int n = n0 + nt * 16 + (lane & 15);
    const float bs = bias ? bias[n] : 0.f;
#pragma unroll
    for (int r = 0; r < 8; ++r) {
      int m = m0 + mrow + r;
      float c = acc[nt][r] + bs;
      if constexpr (EPI == 0) {
        outF[(size_t)m * EE + n] = c;
      } else {
        int b = m >> 10, s = m & (SS - 1);
        int h = n >> 5,  d = n & (DD - 1);
        if constexpr (EPI == 1) {
          if (phase) c *= phase[b * DD + d];              // broadcast hits D axis (H==D)
          outB[(((size_t)(b * HH + h)) * SS + s) * DD + d] = f2bf(c);
        } else {                                          // EPI == 2: V transposed
          outB[(((size_t)(b * HH + h)) * DD + d) * SS + s] = f2bf(c);
        }
      }
    }
  }
}

// ---------------- attention: per (b,h,q-tile of 32 rows) ----------------
// scores = Qh @ Kh^T * SCALE  (WMMA)  -> LDS f32
// softmax in LDS; normalized attn -> d_out (optional) + bf16 LDS
// ctx = attn_bf16 @ V (from Vt, WMMA) -> ctxF (B,S,E) f32 (optional) + ctxB bf16
__global__ __launch_bounds__(64)
void k_attn(const unsigned short* __restrict__ Qh, const unsigned short* __restrict__ Kh,
            const unsigned short* __restrict__ Vt, float* __restrict__ attn_out,
            float* __restrict__ ctxF, unsigned short* __restrict__ ctxB) {
  constexpr int SCS = SS + 16;                            // f32 score row stride
  constexpr int ABS = SS + 32;                            // bf16 attn row stride
  extern __shared__ char smem[];
  float* sc = (float*)smem;
  unsigned short* ab = (unsigned short*)(smem + (size_t)32 * SCS * sizeof(float));
  __shared__ float rowmx[32], rowinv[32];

  const int lane = threadIdx.x & 31;
  const int wave = threadIdx.x >> 5;                      // 2 waves x 16 q-rows
  const int bh   = blockIdx.x >> 5;                       // / (SS/32)
  const int q0   = (blockIdx.x & 31) * 32;
  const int b    = bh >> 5;                               // / HH
  const int h    = bh & 31;

  // ---- scores: 32 rows x 1024 cols ----
  {
    const int qrow = q0 + wave * 16 + (lane & 15);
    const unsigned short* qb = Qh + ((size_t)bh * SS + qrow) * DD + (lane >> 4) * 8;
    v16bf aq = make_frag(*(const u32x4*)qb, *(const u32x4*)(qb + 16));
    for (int nt = 0; nt < SS / 16; ++nt) {
      const unsigned short* kb =
          Kh + ((size_t)bh * SS + nt * 16 + (lane & 15)) * DD + (lane >> 4) * 8;
      v16bf bk = make_frag(*(const u32x4*)kb, *(const u32x4*)(kb + 16));
      v8f s = {};
      s = wmma_bf16(aq, bk, s);
#pragma unroll
      for (int r = 0; r < 8; ++r) {
        int ml = wave * 16 + (lane >> 4) * 8 + r;
        sc[ml * SCS + nt * 16 + (lane & 15)] = s[r] * SCALE;
      }
    }
  }
  __syncthreads();

  // ---- row max / sum (2 threads per row) ----
  {
    const int row = threadIdx.x >> 1;
    const int half = threadIdx.x & 1;
    const float* rp = sc + row * SCS + half * (SS / 2);
    float mx = -3.4e38f;
    for (int c = 0; c < SS / 2; ++c) mx = fmaxf(mx, rp[c]);
    mx = fmaxf(mx, __shfl_xor(mx, 1, 32));
    float sum = 0.f;
    for (int c = 0; c < SS / 2; ++c) sum += __expf(rp[c] - mx);
    sum += __shfl_xor(sum, 1, 32);
    if (half == 0) { rowmx[row] = mx; rowinv[row] = 1.f / sum; }
  }
  __syncthreads();

  // ---- normalize: write f32 attn (coalesced) + bf16 to LDS ----
  for (int i = 0; i < (32 * SS) / 64; ++i) {
    int idx = i * 64 + threadIdx.x;
    int r = idx >> 10;
    int c = idx & (SS - 1);
    float v = __expf(sc[r * SCS + c] - rowmx[r]) * rowinv[r];
    ab[r * ABS + c] = f2bf(v);
    if (attn_out) attn_out[(((size_t)bh * SS) + q0 + r) * SS + c] = v;
  }
  __syncthreads();

  // ---- context: (32 x 1024) @ (1024 x 32) ----
  v8f cacc[2] = {};
  const unsigned short* abase =
      ab + (size_t)(wave * 16 + (lane & 15)) * ABS + (lane >> 4) * 8;
  for (int k0 = 0; k0 < SS; k0 += 32) {
    v16bf af = make_frag(*(const u32x4*)(abase + k0), *(const u32x4*)(abase + k0 + 16));
#pragma unroll
    for (int nt = 0; nt < 2; ++nt) {
      const unsigned short* vb =
          Vt + ((size_t)bh * DD + nt * 16 + (lane & 15)) * SS + k0 + (lane >> 4) * 8;
      v16bf bv = make_frag(*(const u32x4*)vb, *(const u32x4*)(vb + 16));
      cacc[nt] = wmma_bf16(af, bv, cacc[nt]);
    }
  }
#pragma unroll
  for (int nt = 0; nt < 2; ++nt) {
#pragma unroll
    for (int r = 0; r < 8; ++r) {
      int srow = q0 + wave * 16 + (lane >> 4) * 8 + r;
      int d = nt * 16 + (lane & 15);
      float c = cacc[nt][r];
      size_t idx = ((size_t)b * SS + srow) * EE + h * DD + d;
      if (ctxF) ctxF[idx] = c;
      ctxB[idx] = f2bf(c);
    }
  }
}

// ---------------- host orchestration ----------------
extern "C" void kernel_launch(void* const* d_in, const int* in_sizes, int n_in,
                              void* d_out, int out_size, void* d_ws, size_t ws_size,
                              hipStream_t stream) {
  (void)in_sizes; (void)n_in; (void)out_size; (void)ws_size;
  const float* x          = (const float*)d_in[0];
  const float* q_w        = (const float*)d_in[1];
  const float* q_b        = (const float*)d_in[2];
  const float* k_w        = (const float*)d_in[3];
  const float* k_b        = (const float*)d_in[4];
  const float* v_w        = (const float*)d_in[5];
  const float* v_b        = (const float*)d_in[6];
  const float* ph_w1      = (const float*)d_in[7];
  const float* ph_b1      = (const float*)d_in[8];
  const float* ph_w2      = (const float*)d_in[9];
  const float* ph_b2      = (const float*)d_in[10];
  const float* in_proj_w  = (const float*)d_in[11];
  const float* in_proj_b  = (const float*)d_in[12];
  const float* out_proj_w = (const float*)d_in[13];
  const float* out_proj_b = (const float*)d_in[14];
  const float* final_w    = (const float*)d_in[15];
  const float* final_b    = (const float*)d_in[16];

  float* out   = (float*)d_out;                           // (B,S,E)
  float* attn1 = out + (size_t)BB * SS * EE;              // (B,H,S,S)

  char* wp = (char*)d_ws;
  auto alloc = [&](size_t bytes) -> char* {
    char* p = wp; wp += (bytes + 255) & ~((size_t)255); return p;
  };
  unsigned short* xbf  = (unsigned short*)alloc((size_t)TOK * EE * 2);
  unsigned short* qwbf = (unsigned short*)alloc((size_t)EE * EE * 2);
  unsigned short* kwbf = (unsigned short*)alloc((size_t)EE * EE * 2);
  unsigned short* vwbf = (unsigned short*)alloc((size_t)EE * EE * 2);
  unsigned short* ipbf = (unsigned short*)alloc((size_t)3 * EE * EE * 2);
  unsigned short* opbf = (unsigned short*)alloc((size_t)EE * EE * 2);
  unsigned short* fwbf = (unsigned short*)alloc((size_t)EE * EE * 2);
  float* xm    = (float*)alloc((size_t)BB * EE * 4);
  float* hbuf  = (float*)alloc((size_t)BB * EE * 4);
  float* phase = (float*)alloc((size_t)BB * HH * 4);
  unsigned short* Qh  = (unsigned short*)alloc((size_t)TOK * EE * 2);
  unsigned short* Kh  = (unsigned short*)alloc((size_t)TOK * EE * 2);
  unsigned short* Vt  = (unsigned short*)alloc((size_t)TOK * EE * 2);
  float* ctxF  = (float*)alloc((size_t)TOK * EE * 4);
  unsigned short* ctxB = (unsigned short*)alloc((size_t)TOK * EE * 2);
  float* ent   = (float*)alloc((size_t)TOK * EE * 4);
  // reuse freed buffers for phase-2 tensors
  unsigned short* Q2h   = Qh;
  unsigned short* K2h   = Kh;
  unsigned short* V2t   = Vt;
  unsigned short* ctx2B = xbf;     // x_bf16 no longer needed after QKV GEMMs
  unsigned short* sumbf = ctxB;    // ctxB free after in_proj GEMMs

  auto castN = [&](const float* s, unsigned short* d, int n) {
    k_cast_bf16<<<(n / 4 + 255) / 256, 256, 0, stream>>>(s, d, n);
  };
  castN(x, xbf, TOK * EE);
  castN(q_w, qwbf, EE * EE);
  castN(k_w, kwbf, EE * EE);
  castN(v_w, vwbf, EE * EE);
  castN(in_proj_w, ipbf, 3 * EE * EE);
  castN(out_proj_w, opbf, EE * EE);
  castN(final_w, fwbf, EE * EE);

  // phase path (tiny, f32 VALU)
  k_colmean<<<(BB * EE) / 256, 256, 0, stream>>>(x, xm);
  k_dense_act<<<(BB * EE + 127) / 128, 128, 0, stream>>>(xm, ph_w1, ph_b1, hbuf,
                                                         BB, EE, EE, 0);
  k_dense_act<<<(BB * HH + 127) / 128, 128, 0, stream>>>(hbuf, ph_w2, ph_b2, phase,
                                                         BB, HH, EE, 1);

  dim3 gg(TOK / 64, EE / 64);
  // Q/K with phase, V transposed
  k_gemm<1><<<gg, 128, 0, stream>>>(xbf, qwbf, q_b, nullptr, Qh, phase);
  k_gemm<1><<<gg, 128, 0, stream>>>(xbf, kwbf, k_b, nullptr, Kh, phase);
  k_gemm<2><<<gg, 128, 0, stream>>>(xbf, vwbf, v_b, nullptr, Vt, nullptr);

  const size_t ATTN_LDS = (size_t)32 * (SS + 16) * 4 + (size_t)32 * (SS + 32) * 2;
  k_attn<<<BB * HH * (SS / 32), 64, ATTN_LDS, stream>>>(Qh, Kh, Vt, attn1, ctxF, ctxB);

  // second MHA: in_proj slices
  k_gemm<1><<<gg, 128, 0, stream>>>(ctxB, ipbf,               in_proj_b,          nullptr, Q2h, nullptr);
  k_gemm<1><<<gg, 128, 0, stream>>>(ctxB, ipbf + (size_t)EE * EE,     in_proj_b + EE,     nullptr, K2h, nullptr);
  k_gemm<2><<<gg, 128, 0, stream>>>(ctxB, ipbf + (size_t)2 * EE * EE, in_proj_b + 2 * EE, nullptr, V2t, nullptr);

  k_attn<<<BB * HH * (SS / 32), 64, ATTN_LDS, stream>>>(Q2h, K2h, V2t, nullptr, nullptr, ctx2B);

  k_gemm<0><<<gg, 128, 0, stream>>>(ctx2B, opbf, out_proj_b, ent, nullptr, nullptr);
  k_add_cast<<<(TOK * EE) / 256, 256, 0, stream>>>(ent, ctxF, sumbf, TOK * EE);
  k_gemm<0><<<gg, 128, 0, stream>>>(sumbf, fwbf, final_b, out, nullptr, nullptr);
}
